// CodebookLoRASTELinear_43808666419597
// MI455X (gfx1250) — compile-verified, matrix operand order checked by reading, not run on hardware
//
#include <hip/hip_runtime.h>
#include <hip/hip_bf16.h>

typedef __attribute__((ext_vector_type(16))) __bf16 v16bf;
typedef __attribute__((ext_vector_type(8)))  __bf16 v8bf;
typedef __attribute__((ext_vector_type(8)))  float  v8f;

#define O_DIM 4096
#define I_DIM 4096
#define RANK  64
#define GROUPSZ 128
#define ALPHA_OVER_RANK 0.5f   // LORA_ALPHA / LORA_RANK = 32/64

#define BM 128
#define BN 128
#define BK 64
#define LDK 72   // padded LDS row pitch (bf16 elems); 72*2=144 B, 16B aligned

union Frag { v16bf v; v8bf h[2]; };

__device__ __forceinline__ void wait_async0() {
#if __has_builtin(__builtin_amdgcn_s_wait_asynccnt)
    __builtin_amdgcn_s_wait_asynccnt(0);
#else
    asm volatile("s_wait_asynccnt 0x0" ::: "memory");
#endif
}

// Async global -> LDS, 16 bytes per lane, no VGPR transit (ASYNCcnt-tracked).
__device__ __forceinline__ void async_copy_b128(unsigned lds_off, const void* gptr) {
    asm volatile("global_load_async_to_lds_b128 %0, %1, off"
                 :: "v"(lds_off), "v"((unsigned long long)gptr)
                 : "memory");
}

// ---------------------------------------------------------------------------
// Kernel 1: fused LoRA-delta + groupwise codebook quantize -> bf16 w_eff[O][I]
// ---------------------------------------------------------------------------
__global__ __launch_bounds__(256) void build_weff(
    const float* __restrict__ weight, const float* __restrict__ scale_log,
    const float* __restrict__ codebook, const float* __restrict__ loraA,
    const float* __restrict__ loraB, __bf16* __restrict__ weff)
{
    __shared__ float sB[RANK];
    const int o     = blockIdx.y;
    const int iBase = blockIdx.x * 2048 + threadIdx.x * 8;
    if (threadIdx.x < RANK) sB[threadIdx.x] = loraB[o * RANK + threadIdx.x];
    __syncthreads();

    float acc[8];
    #pragma unroll
    for (int e = 0; e < 8; ++e) acc[e] = 0.f;
    for (int r = 0; r < RANK; ++r) {
        const float b = sB[r];
        const float* a = loraA + (size_t)r * I_DIM + iBase;
        #pragma unroll
        for (int e = 0; e < 8; ++e) acc[e] = fmaf(b, a[e], acc[e]);
    }

    float c0 = codebook[0], c1 = codebook[1], c2 = codebook[2], c3 = codebook[3];
    const float cbmax = fmaxf(fmaxf(fabsf(c0), fabsf(c1)), fmaxf(fabsf(c2), fabsf(c3)));
    const float inv = 1.f / fmaxf(cbmax, 1e-8f);
    c0 *= inv; c1 *= inv; c2 *= inv; c3 *= inv;
    const float t0 = 0.5f * (c0 + c1), t1 = 0.5f * (c1 + c2), t2 = 0.5f * (c2 + c3);

    const float scale = __expf(scale_log[o * (I_DIM / GROUPSZ) + iBase / GROUPSZ]);
    const float rscale = 1.f / scale;

    v8bf outv;
    #pragma unroll
    for (int e = 0; e < 8; ++e) {
        const float delta = acc[e] * ALPHA_OVER_RANK;
        const float n = (weight[(size_t)o * I_DIM + iBase + e] + delta) * rscale;
        const int idx = (int)(n > t0) + (int)(n > t1) + (int)(n > t2);
        const float q = (idx == 0) ? c0 : (idx == 1) ? c1 : (idx == 2) ? c2 : c3;
        outv[e] = (__bf16)(q * scale + delta);
    }
    *(v8bf*)(weff + (size_t)o * I_DIM + iBase) = outv;
}

// ---------------------------------------------------------------------------
// Kernel 2: x f32 -> bf16 (one-time; GEMM operands then total 96 MB -> L2-resident)
// ---------------------------------------------------------------------------
__global__ __launch_bounds__(256) void cvt_x(const float* __restrict__ x,
                                             __bf16* __restrict__ xb, int n)
{
    const int i = (blockIdx.x * 256 + threadIdx.x) * 8;
    if (i >= n) return;
    v8bf o;
    #pragma unroll
    for (int e = 0; e < 8; ++e) o[e] = (__bf16)x[i + e];
    *(v8bf*)(xb + i) = o;
}

// ---------------------------------------------------------------------------
// Kernel 3: y[M][N] = A[M][K] * B[N][K]^T, bf16 in / f32 accumulate.
// 8 wave32 waves; block tile 128x128; wave tile 64x32 (4x2 WMMA accumulators).
// Double-buffered LDS fed by GLOBAL_LOAD_ASYNC_TO_LDS_B128 (ASYNCcnt).
// ---------------------------------------------------------------------------
__global__ __launch_bounds__(256) void gemm_bf16(
    const __bf16* __restrict__ A,   // [M][K] bf16
    const __bf16* __restrict__ B,   // [N][K] bf16 (row n of w_eff = B-matrix col n)
    float* __restrict__ C, int M, int N, int K)
{
    // smem[buf][0] = A tile, smem[buf][1] = B tile
    __shared__ __bf16 smem[2][2][BM][LDK];

    const int tid  = threadIdx.x;
    const int lane = tid & 31;
    const int wave = tid >> 5;
    const int m0 = blockIdx.y * BM;
    const int n0 = blockIdx.x * BN;
    const int wm = (wave >> 2) * 64;   // wave M offset: 0 / 64
    const int wn = (wave & 3) * 32;    // wave N offset: 0 / 32 / 64 / 96
    const int lrow = lane & 15;        // matrix row (A) / col (B,D)
    const int lhi  = lane >> 4;        // K sub-chunk selector

    const unsigned lds_base = (unsigned)(size_t)(&smem[0][0][0][0]);
    const int row_s = tid >> 1;            // staging row for this thread (2 thr/row)
    const int ch_s  = (tid & 1) * 32;      // staging col: 0 or 32 (4 chunks of 8 each)
    const unsigned offA0 = lds_base + (unsigned)((0 * 2 + 0) * BM * LDK + row_s * LDK + ch_s) * 2u;
    const unsigned offB0 = lds_base + (unsigned)((0 * 2 + 1) * BM * LDK + row_s * LDK + ch_s) * 2u;
    const unsigned bufStride = (unsigned)(2 * BM * LDK) * 2u;   // bytes between buffers
    const __bf16* gA0 = A + (size_t)(m0 + row_s) * K + ch_s;
    const __bf16* gB0 = B + (size_t)(n0 + row_s) * K + ch_s;

    // Stage one 128x64 tile pair: each thread issues 4+4 async b128 copies
    auto stage = [&](int ibuf, int kc) {
        #pragma unroll
        for (int c = 0; c < 4; ++c) {
            async_copy_b128(offA0 + (unsigned)ibuf * bufStride + c * 16u, gA0 + kc + c * 8);
            async_copy_b128(offB0 + (unsigned)ibuf * bufStride + c * 16u, gB0 + kc + c * 8);
        }
    };

    v8f acc[4][2];
    #pragma unroll
    for (int i = 0; i < 4; ++i)
        #pragma unroll
        for (int j = 0; j < 2; ++j) acc[i][j] = (v8f){};

    stage(0, 0);
    int ibuf = 0;
    for (int kc = 0; kc < K; kc += BK) {
        wait_async0();       // this wave's copies into smem[ibuf] complete
        __syncthreads();     // all waves' copies visible; prev buffer free to overwrite

        if (kc + BK < K) stage(ibuf ^ 1, kc + BK);   // overlap next copies with WMMAs

        #pragma unroll
        for (int ks = 0; ks < BK; ks += 32) {
            Frag fa[4], fb[2];
            #pragma unroll
            for (int t = 0; t < 4; ++t) {   // A: lane = M row; K chunks 8*lhi, +16
                fa[t].h[0] = *(const v8bf*)&smem[ibuf][0][wm + t * 16 + lrow][ks + 8 * lhi];
                fa[t].h[1] = *(const v8bf*)&smem[ibuf][0][wm + t * 16 + lrow][ks + 8 * lhi + 16];
            }
            #pragma unroll
            for (int t = 0; t < 2; ++t) {   // B: lane = N col (row of w_eff)
                fb[t].h[0] = *(const v8bf*)&smem[ibuf][1][wn + t * 16 + lrow][ks + 8 * lhi];
                fb[t].h[1] = *(const v8bf*)&smem[ibuf][1][wn + t * 16 + lrow][ks + 8 * lhi + 16];
            }
            #pragma unroll
            for (int i = 0; i < 4; ++i)
                #pragma unroll
                for (int j = 0; j < 2; ++j)
                    acc[i][j] = __builtin_amdgcn_wmma_f32_16x16x32_bf16(
                        false, fa[i].v, false, fb[j].v,
                        (short)0, acc[i][j], false, false);
        }
        ibuf ^= 1;
    }

    // D layout: VGPR v -> row v + 8*lhi within 16x16 tile; lane%16 -> col
    #pragma unroll
    for (int i = 0; i < 4; ++i) {
        const int mbase = m0 + wm + i * 16 + 8 * lhi;
        #pragma unroll
        for (int j = 0; j < 2; ++j) {
            const int nn = n0 + wn + j * 16 + lrow;
            #pragma unroll
            for (int v = 0; v < 8; ++v)
                C[(size_t)(mbase + v) * N + nn] = acc[i][j][v];
        }
    }
}

// ---------------------------------------------------------------------------
extern "C" void kernel_launch(void* const* d_in, const int* in_sizes, int n_in,
                              void* d_out, int out_size, void* d_ws, size_t ws_size,
                              hipStream_t stream) {
    const float* x         = (const float*)d_in[0];
    const float* weight    = (const float*)d_in[1];
    const float* scale_log = (const float*)d_in[2];
    const float* codebook  = (const float*)d_in[3];
    const float* loraA     = (const float*)d_in[4];
    const float* loraB     = (const float*)d_in[5];
    float* out = (float*)d_out;

    const int I = I_DIM, O = O_DIM;
    const int M = in_sizes[0] / I;    // 4 * 2048 = 8192

    __bf16* weff = (__bf16*)d_ws;                                    // 32 MB
    __bf16* xb   = (__bf16*)((char*)d_ws + (size_t)O * I * 2);       // 64 MB

    dim3 g1(I / 2048, O);
    build_weff<<<g1, 256, 0, stream>>>(weight, scale_log, codebook, loraA, loraB, weff);

    const int nx = M * I;
    cvt_x<<<(nx / 8 + 255) / 256, 256, 0, stream>>>(x, xb, nx);

    dim3 g3(O / BN, M / BM);
    gemm_bf16<<<g3, 256, 0, stream>>>(xb, weff, out, M, O, I);
}
